// SwinTransformerEncoderLayer_57045755625848
// MI455X (gfx1250) — matrix operand order repaired and checked by reading, not compiled, not run on hardware
//
#include <hip/hip_runtime.h>

#define TOK   65536
#define EDIM  512
#define NHEAD 16
#define HD    32
#define NWIN  1024
#define DFF   2048
#define EPS   1e-5f

typedef __attribute__((ext_vector_type(16))) _Float16 v16h;
typedef __attribute__((ext_vector_type(8)))  float    v8f;
typedef unsigned short u16;
typedef unsigned int   u32;

union Frag { v16h v; u32 u[8]; _Float16 h[16]; };

__device__ inline v8f wmma_f16(v16h a, v16h b, v8f c) {
  return __builtin_amdgcn_wmma_f32_16x16x32_f16(false, a, false, b, (short)0, c, false, false);
}

// Async global->LDS copy (16B per lane), tracked by ASYNCcnt (CDNA5 ISA).
__device__ inline void async_ld_b128(u32 lds_off, const void* gsrc) {
  asm volatile("global_load_async_to_lds_b128 %0, %1, off"
               :: "v"(lds_off), "v"((unsigned long long)(uintptr_t)gsrc)
               : "memory");
}
__device__ inline void wait_async() {
  asm volatile("s_wait_asynccnt 0x0" ::: "memory");
}

// A-fragment (16x32 f16) from a row-major half buffer, leading dim ld (halves).
// ISA layout: lane r=lane&15 holds row r; VGPR v holds K = (v/4)*16 + (v%4)*2 + 8*(lane>>4).
__device__ inline v16h load_a(const u16* base, int ld, int lane) {
  const int r = lane & 15, hh = lane >> 4;
  Frag f;
#pragma unroll
  for (int v = 0; v < 8; ++v) {
    int k = ((v >> 2) << 4) + ((v & 3) << 1) + (hh << 3);
    f.u[v] = *(const u32*)(base + r * ld + k);
  }
  return f.v;
}

// B-fragment (32x16 f16): base points at [n0][k0] of row-major [N][K] buffer (ld=K halves).
// lane col = lane&15; VGPR v holds K = (v/4)*16 + (v%4)*2 + 8*(lane>>4) (same as A).
__device__ inline v16h load_b(const u16* base, int ld, int lane) {
  const u16* p = base + (size_t)(lane & 15) * ld + ((lane >> 4) << 3);
  Frag f;
#pragma unroll
  for (int v = 0; v < 8; ++v) f.u[v] = *(const u32*)(p + ((v >> 2) << 4) + ((v & 3) << 1));
  return f.v;
}

__device__ inline float redmax16(float x) {
#pragma unroll
  for (int m = 1; m < 16; m <<= 1) x = fmaxf(x, __shfl_xor(x, m, 32));
  return x;
}
__device__ inline float redsum16(float x) {
#pragma unroll
  for (int m = 1; m < 16; m <<= 1) x += __shfl_xor(x, m, 32);
  return x;
}

__device__ inline int tok_of(int win, int n) {
  int b = win >> 8, wr = (win >> 4) & 15, wc = win & 15;
  int row = (wr << 3) + (n >> 3), col = (wc << 3) + (n & 7);
  return (b << 14) + (row << 7) + col;
}

__device__ inline u16 to_h(float x) { _Float16 h = (_Float16)x; return *(u16*)&h; }

// ---------------- converts ----------------
__global__ void k_cvt(const float* __restrict__ in, u16* __restrict__ out, int n) {
  int i = blockIdx.x * blockDim.x + threadIdx.x;
  if (i < n) out[i] = to_h(in[i]);
}

__global__ void k_bias(const float* __restrict__ pe, float* __restrict__ bias) {
  int i = blockIdx.x * blockDim.x + threadIdx.x;
  if (i < 4096) {
    int q = i >> 6, k = i & 63;
    int qh = q >> 3, qw = q & 7, kh = k >> 3, kw = k & 7;
    bias[i] = pe[(kh - qh + 7) * 15 + (kw - qw + 7)];
  }
}

// ---------------- QKV projection: M=65536, N=1536, K=512 ----------------
// 128x64 tile per block; output scattered into per-window layout:
//   win base = win*98304 halves: Q[n][512] @+0, K[n][512] @+32768, V^T[dim][64] @+65536
__global__ __launch_bounds__(256) void k_qkv(const u16* __restrict__ xh,
                                             const u16* __restrict__ wq,
                                             const float* __restrict__ bqkv,
                                             u16* __restrict__ qkvh) {
  __shared__ __align__(16) u16 la[128 * 32];
  const int tid = threadIdx.x, lane = tid & 31, wid = tid >> 5;
  const int bx = blockIdx.x, by = blockIdx.y;
  const int wm = wid >> 1, wn = wid & 1;
  const u32 lab = (u32)(uintptr_t)la;
  v8f acc[2][2] = {};
  for (int ks = 0; ks < 16; ++ks) {
    __syncthreads();
    // async-stage A tile (128x32 halves = 8KB) straight into LDS, 2x16B per lane
#pragma unroll
    for (int i = 0; i < 2; ++i) {
      int c = i * 256 + tid;              // 512 x 16B chunks
      int row = c >> 2, seg = c & 3;
      async_ld_b128(lab + c * 16,
                    xh + (size_t)(bx * 128 + row) * EDIM + ks * 32 + seg * 8);
    }
    wait_async();
    __syncthreads();
#pragma unroll
    for (int ml = 0; ml < 2; ++ml) {
      v16h af = load_a(la + (wm * 32 + ml * 16) * 32, 32, lane);
#pragma unroll
      for (int jl = 0; jl < 2; ++jl) {
        int nb = by * 64 + wn * 32 + jl * 16;
        v16h bf = load_b(wq + (size_t)nb * EDIM + ks * 32, EDIM, lane);
        acc[ml][jl] = wmma_f16(af, bf, acc[ml][jl]);
      }
    }
  }
  const int h8 = (lane >> 4) << 3, c0 = lane & 15;
#pragma unroll
  for (int ml = 0; ml < 2; ++ml)
#pragma unroll
    for (int jl = 0; jl < 2; ++jl)
#pragma unroll
      for (int v = 0; v < 8; ++v) {
        int row = wm * 32 + ml * 16 + h8 + v;
        int col = by * 64 + wn * 32 + jl * 16 + c0;
        int t = bx * 128 + row;
        float val = acc[ml][jl][v] + bqkv[col];
        int b = t >> 14, pos = t & 16383, pr = pos >> 7, pc = pos & 127;
        int win = (b << 8) + ((pr >> 3) << 4) + (pc >> 3);
        int n = ((pr & 7) << 3) + (pc & 7);
        size_t base = (size_t)win * 98304;
        size_t dst;
        if (col < 512)       dst = base + (size_t)n * 512 + col;
        else if (col < 1024) dst = base + 32768 + (size_t)n * 512 + (col - 512);
        else                 dst = base + 65536 + (size_t)(col - 1024) * 64 + n;
        qkvh[dst] = to_h(val);
      }
}

// ---------------- windowed attention: 1 block (4 waves) per window ----------------
__global__ __launch_bounds__(128) void k_attn(const u16* __restrict__ qkvh,
                                              const float* __restrict__ bias,
                                              u16* __restrict__ attnh) {
  __shared__ __align__(16) u16 pbuf[4 * 64 * 64];   // 32 KB, 8 KB per wave
  const int tid = threadIdx.x, lane = tid & 31, wid = tid >> 5;
  const int win = blockIdx.x;
  const u16* qb = qkvh + (size_t)win * 98304;
  const u16* kb = qb + 32768;
  const u16* vb = qb + 65536;
  u16* p = pbuf + wid * 4096;
  const int hh = lane >> 4, c0 = lane & 15, h8 = hh << 3;
  const float scale = 0.17677669529663687f;   // 1/sqrt(32)
  for (int hq = 0; hq < 4; ++hq) {
    const int head = wid * 4 + hq;
    const int hoff = head * HD;
    // ---- S = Q K^T : 4x4 tiles, K=32 in one WMMA step ----
    v8f s[4][4] = {};
    v16h qf[4], kf[4];
#pragma unroll
    for (int mi = 0; mi < 4; ++mi) qf[mi] = load_a(qb + (size_t)(mi * 16) * EDIM + hoff, EDIM, lane);
#pragma unroll
    for (int jn = 0; jn < 4; ++jn) kf[jn] = load_b(kb + (size_t)(jn * 16) * EDIM + hoff, EDIM, lane);
#pragma unroll
    for (int mi = 0; mi < 4; ++mi)
#pragma unroll
      for (int jn = 0; jn < 4; ++jn)
        s[mi][jn] = wmma_f16(qf[mi], kf[jn], s[mi][jn]);
    // ---- bias + scale + row softmax (rows per lane-half, cols across 16 lanes x 4 tiles) ----
#pragma unroll
    for (int mi = 0; mi < 4; ++mi)
#pragma unroll
      for (int v = 0; v < 8; ++v) {
        int row = mi * 16 + h8 + v;
        float e0[4];
        float m = -3.0e38f;
#pragma unroll
        for (int jn = 0; jn < 4; ++jn) {
          float x = s[mi][jn][v] * scale + bias[row * 64 + jn * 16 + c0];
          e0[jn] = x;
          m = fmaxf(m, x);
        }
        m = redmax16(m);
        float sum = 0.f;
#pragma unroll
        for (int jn = 0; jn < 4; ++jn) { e0[jn] = __expf(e0[jn] - m); sum += e0[jn]; }
        sum = redsum16(sum);
        float inv = __frcp_rn(sum);
#pragma unroll
        for (int jn = 0; jn < 4; ++jn)
          p[row * 64 + jn * 16 + c0] = to_h(e0[jn] * inv);
      }
    // ---- O = P V : M=64, K=64 (2 steps), N=32 ; V stored transposed [dim][tok] ----
    v8f o[4][2] = {};
#pragma unroll
    for (int ks = 0; ks < 2; ++ks) {
      v16h bf[2];
#pragma unroll
      for (int jn = 0; jn < 2; ++jn)
        bf[jn] = load_b(vb + (size_t)(hoff + jn * 16) * 64 + ks * 32, 64, lane);
#pragma unroll
      for (int mi = 0; mi < 4; ++mi) {
        v16h af = load_a(p + mi * 16 * 64 + ks * 32, 64, lane);
#pragma unroll
        for (int jn = 0; jn < 2; ++jn)
          o[mi][jn] = wmma_f16(af, bf[jn], o[mi][jn]);
      }
    }
#pragma unroll
    for (int mi = 0; mi < 4; ++mi)
#pragma unroll
      for (int jn = 0; jn < 2; ++jn)
#pragma unroll
        for (int v = 0; v < 8; ++v) {
          int n = mi * 16 + h8 + v;
          int col = hoff + jn * 16 + c0;
          attnh[((size_t)win * 64 + n) * EDIM + col] = to_h(o[mi][jn][v]);
        }
  }
}

// ---------------- out-proj + residual + LN1 (block = 1 window, full N=512) ----------------
__global__ __launch_bounds__(256) void k_oproj_ln1(const u16* __restrict__ attnh,
                                                   const u16* __restrict__ wo,
                                                   const float* __restrict__ bo,
                                                   const float* __restrict__ x,
                                                   const float* __restrict__ g,
                                                   const float* __restrict__ bt,
                                                   float* __restrict__ zf,
                                                   u16* __restrict__ zh) {
  __shared__ __align__(16) u16 la[64 * 32];
  __shared__ float ssum[64], ssq[64];
  const int tid = threadIdx.x, lane = tid & 31, wid = tid >> 5;
  const int win = blockIdx.x;
  const u32 lab = (u32)(uintptr_t)la;
  v8f acc[4][4] = {};
  for (int ks = 0; ks < 16; ++ks) {
    __syncthreads();
    {   // async-stage A tile (64x32 halves = 4KB), 16B per lane
      int row = tid >> 2, seg = tid & 3;
      async_ld_b128(lab + tid * 16,
                    attnh + ((size_t)win * 64 + row) * EDIM + ks * 32 + seg * 8);
    }
    wait_async();
    __syncthreads();
#pragma unroll
    for (int mi = 0; mi < 4; ++mi) {
      v16h af = load_a(la + mi * 16 * 32, 32, lane);
#pragma unroll
      for (int jn = 0; jn < 4; ++jn) {
        v16h bf = load_b(wo + (size_t)(wid * 64 + jn * 16) * EDIM + ks * 32, EDIM, lane);
        acc[mi][jn] = wmma_f16(af, bf, acc[mi][jn]);
      }
    }
  }
  const int hh = lane >> 4, c0 = lane & 15, h8 = hh << 3;
  if (tid < 64) { ssum[tid] = 0.f; ssq[tid] = 0.f; }
  __syncthreads();
#pragma unroll
  for (int mi = 0; mi < 4; ++mi)
#pragma unroll
    for (int v = 0; v < 8; ++v) {
      int r = mi * 16 + h8 + v;
      int t = tok_of(win, r);
      float psum = 0.f, psq = 0.f;
#pragma unroll
      for (int jn = 0; jn < 4; ++jn) {
        int col = wid * 64 + jn * 16 + c0;
        float val = acc[mi][jn][v] + bo[col] + x[(size_t)t * EDIM + col];
        acc[mi][jn][v] = val;
        psum += val; psq += val * val;
      }
      psum = redsum16(psum); psq = redsum16(psq);
      if (c0 == 0) { atomicAdd(&ssum[r], psum); atomicAdd(&ssq[r], psq); }
    }
  __syncthreads();
#pragma unroll
  for (int mi = 0; mi < 4; ++mi)
#pragma unroll
    for (int v = 0; v < 8; ++v) {
      int r = mi * 16 + h8 + v;
      int t = tok_of(win, r);
      float mean = ssum[r] * (1.f / 512.f);
      float var  = ssq[r] * (1.f / 512.f) - mean * mean;
      float rs = __frsqrt_rn(var + EPS);
#pragma unroll
      for (int jn = 0; jn < 4; ++jn) {
        int col = wid * 64 + jn * 16 + c0;
        float zv = (acc[mi][jn][v] - mean) * rs * g[col] + bt[col];
        zf[(size_t)t * EDIM + col] = zv;
        zh[(size_t)t * EDIM + col] = to_h(zv);
      }
    }
}

// ---------------- fused FFN (512->2048->512) + residual + LN2 ----------------
__global__ __launch_bounds__(256) void k_ffn_ln2(const u16* __restrict__ zh,
                                                 const float* __restrict__ zf,
                                                 const u16* __restrict__ w1,
                                                 const float* __restrict__ b1,
                                                 const u16* __restrict__ w2,
                                                 const float* __restrict__ b2,
                                                 const float* __restrict__ g,
                                                 const float* __restrict__ bt,
                                                 float* __restrict__ out) {
  __shared__ __align__(16) u16 la[64 * 32];
  __shared__ __align__(16) u16 sh[64 * 64];
  __shared__ float ssum[64], ssq[64];
  const int tid = threadIdx.x, lane = tid & 31, wid = tid >> 5;
  const size_t t0 = (size_t)blockIdx.x * 64;
  const int hh = lane >> 4, c0 = lane & 15, h8 = hh << 3;
  const int g1m = wid >> 1, g1j = (wid & 1) << 1;
  const u32 lab = (u32)(uintptr_t)la;
  v8f o[4][4] = {};
  for (int ch = 0; ch < 32; ++ch) {
    // gemm1: S(64x64) = Z(64x512) @ W1_chunk^T
    v8f s[2] = {};
    for (int ks = 0; ks < 16; ++ks) {
      __syncthreads();
      {   // async-stage Z tile (64x32 halves = 4KB), 16B per lane
        int row = tid >> 2, seg = tid & 3;
        async_ld_b128(lab + tid * 16,
                      zh + (t0 + row) * EDIM + ks * 32 + seg * 8);
      }
      wait_async();
      __syncthreads();
      v16h af = load_a(la + g1m * 16 * 32, 32, lane);
#pragma unroll
      for (int jl = 0; jl < 2; ++jl) {
        v16h bf = load_b(w1 + (size_t)(ch * 64 + (g1j + jl) * 16) * EDIM + ks * 32, EDIM, lane);
        s[jl] = wmma_f16(af, bf, s[jl]);
      }
    }
    __syncthreads();           // prior gemm2 reads of sh complete
#pragma unroll
    for (int jl = 0; jl < 2; ++jl)
#pragma unroll
      for (int v = 0; v < 8; ++v) {
        int row = g1m * 16 + h8 + v;
        int col = (g1j + jl) * 16 + c0;
        float hv = s[jl][v] + b1[ch * 64 + col];
        sh[row * 64 + col] = to_h(fmaxf(hv, 0.f));
      }
    __syncthreads();
    // gemm2: O(64x512) += relu(S) @ W2_chunk ; W2 row n contiguous along dff
#pragma unroll
    for (int ks = 0; ks < 2; ++ks) {
      v16h bf[4];
#pragma unroll
      for (int jn = 0; jn < 4; ++jn)
        bf[jn] = load_b(w2 + (size_t)(wid * 64 + jn * 16) * DFF + ch * 64 + ks * 32, DFF, lane);
#pragma unroll
      for (int mi = 0; mi < 4; ++mi) {
        v16h af = load_a(sh + mi * 16 * 64 + ks * 32, 64, lane);
#pragma unroll
        for (int jn = 0; jn < 4; ++jn)
          o[mi][jn] = wmma_f16(af, bf[jn], o[mi][jn]);
      }
    }
  }
  // epilogue: bias + residual + LN2 -> final output
  if (tid < 64) { ssum[tid] = 0.f; ssq[tid] = 0.f; }
  __syncthreads();
#pragma unroll
  for (int mi = 0; mi < 4; ++mi)
#pragma unroll
    for (int v = 0; v < 8; ++v) {
      int r = mi * 16 + h8 + v;
      size_t t = t0 + r;
      float psum = 0.f, psq = 0.f;
#pragma unroll
      for (int jn = 0; jn < 4; ++jn) {
        int col = wid * 64 + jn * 16 + c0;
        float val = o[mi][jn][v] + b2[col] + zf[t * EDIM + col];
        o[mi][jn][v] = val;
        psum += val; psq += val * val;
      }
      psum = redsum16(psum); psq = redsum16(psq);
      if (c0 == 0) { atomicAdd(&ssum[r], psum); atomicAdd(&ssq[r], psq); }
    }
  __syncthreads();
#pragma unroll
  for (int mi = 0; mi < 4; ++mi)
#pragma unroll
    for (int v = 0; v < 8; ++v) {
      int r = mi * 16 + h8 + v;
      size_t t = t0 + r;
      float mean = ssum[r] * (1.f / 512.f);
      float var  = ssq[r] * (1.f / 512.f) - mean * mean;
      float rs = __frsqrt_rn(var + EPS);
#pragma unroll
      for (int jn = 0; jn < 4; ++jn) {
        int col = wid * 64 + jn * 16 + c0;
        out[t * EDIM + col] = (o[mi][jn][v] - mean) * rs * g[col] + bt[col];
      }
    }
}

extern "C" void kernel_launch(void* const* d_in, const int* in_sizes, int n_in,
                              void* d_out, int out_size, void* d_ws, size_t ws_size,
                              hipStream_t stream) {
  const float* x    = (const float*)d_in[0];
  const float* wqkv = (const float*)d_in[1];
  const float* bqkv = (const float*)d_in[2];
  const float* wout = (const float*)d_in[3];
  const float* bout = (const float*)d_in[4];
  const float* pe   = (const float*)d_in[5];
  const float* w1   = (const float*)d_in[6];
  const float* b1   = (const float*)d_in[7];
  const float* w2   = (const float*)d_in[8];
  const float* b2   = (const float*)d_in[9];
  const float* g1   = (const float*)d_in[10];
  const float* be1  = (const float*)d_in[11];
  const float* g2   = (const float*)d_in[12];
  const float* be2  = (const float*)d_in[13];

  char* ws = (char*)d_ws;
  u16*   xh    = (u16*)(ws);                        //  64 MB  x f16
  u16*   qkvh  = (u16*)(ws + 67108864ull);          // 192 MB  windowed Q/K/V^T f16
  u16*   attnh = (u16*)(ws + 268435456ull);         //  64 MB  attention out f16 (window order)
  float* zf    = (float*)(ws + 335544320ull);       // 128 MB  z f32
  u16*   zh    = (u16*)(ws + 469762048ull);         //  64 MB  z f16
  u16*   whq   = (u16*)(ws + 536870912ull);         // 1.5 MB
  u16*   who   = (u16*)(ws + 538443776ull);         // 0.5 MB
  u16*   wh1   = (u16*)(ws + 538968064ull);         //   2 MB
  u16*   wh2   = (u16*)(ws + 541065216ull);         //   2 MB
  float* bias  = (float*)(ws + 543162368ull);       //  16 KB

  k_cvt<<<131072, 256, 0, stream>>>(x, xh, TOK * EDIM);
  k_cvt<<<3072,   256, 0, stream>>>(wqkv, whq, 3 * EDIM * EDIM);
  k_cvt<<<1024,   256, 0, stream>>>(wout, who, EDIM * EDIM);
  k_cvt<<<4096,   256, 0, stream>>>(w1, wh1, DFF * EDIM);
  k_cvt<<<4096,   256, 0, stream>>>(w2, wh2, EDIM * DFF);
  k_bias<<<16,    256, 0, stream>>>(pe, bias);

  dim3 gq(512, 24);
  k_qkv<<<gq, 256, 0, stream>>>(xh, whq, bqkv, qkvh);
  k_attn<<<NWIN, 128, 0, stream>>>(qkvh, bias, attnh);
  k_oproj_ln1<<<NWIN, 256, 0, stream>>>(attnh, who, bout, x, g1, be1, zf, zh);
  k_ffn_ln2<<<TOK / 64, 256, 0, stream>>>(zh, zf, wh1, b1, wh2, b2, g2, be2, (float*)d_out);
}
